// AdjGCN_69329362092560
// MI455X (gfx1250) — matrix operand order repaired and checked by reading, not compiled, not run on hardware
//
#include <hip/hip_runtime.h>

// ---------------------------------------------------------------------------
// Types for CDNA5 WMMA (wave32, 16x16x32 bf16 -> f32)
// ---------------------------------------------------------------------------
typedef __attribute__((ext_vector_type(16))) __bf16 v16bf;
typedef __attribute__((ext_vector_type(8)))  __bf16 v8bf;
typedef __attribute__((ext_vector_type(8)))  float  v8f;

union ABFrag {
    v16bf v;
    v8bf  h[2];
};

__device__ __forceinline__ __bf16 to_bf16(float f) { return (__bf16)f; }

// ---------------------------------------------------------------------------
// Degree / normalization kernels
// ---------------------------------------------------------------------------
__global__ void init_deg_kernel(float* __restrict__ deg, int n) {
    int i = blockIdx.x * blockDim.x + threadIdx.x;
    if (i < n) deg[i] = 1.0f;  // self loop contributes 1
}

__global__ void count_deg_kernel(const long long* __restrict__ rows,
                                 float* __restrict__ deg, int E) {
    int e = blockIdx.x * blockDim.x + threadIdx.x;
    if (e < E) atomicAdd(&deg[(int)rows[e]], 1.0f);
}

__global__ void edge_weight_kernel(const long long* __restrict__ rows,
                                   const long long* __restrict__ cols,
                                   const float* __restrict__ deg,
                                   float* __restrict__ wn, int E, int N) {
    int e = blockIdx.x * blockDim.x + threadIdx.x;
    if (e >= E + N) return;
    int r, c;
    if (e < E) { r = (int)rows[e]; c = (int)cols[e]; }
    else       { r = c = e - E; }
    wn[e] = __frsqrt_rn(deg[r]) * __frsqrt_rn(deg[c]);
}

// ---------------------------------------------------------------------------
// f32 -> bf16 conversion (vectorized)
// ---------------------------------------------------------------------------
__global__ void cvt_bf16_kernel(const float* __restrict__ in,
                                __bf16* __restrict__ out, size_t n4) {
    size_t i = ((size_t)blockIdx.x * blockDim.x + threadIdx.x);
    if (i >= n4) return;
    size_t b = i * 4;
    float4 v = *(const float4*)(in + b);
    out[b + 0] = to_bf16(v.x);
    out[b + 1] = to_bf16(v.y);
    out[b + 2] = to_bf16(v.z);
    out[b + 3] = to_bf16(v.w);
}

// ---------------------------------------------------------------------------
// Pack W (K x Nout, row-major f32) into WMMA B-fragment layout (bf16).
// Fragment for tile (nb 16-cols, kb 32-deep): per lane L a contiguous run of
// 16 bf16: n = nb*16 + (L&15), k = kb*32 + (L>>4)*16 + j  (j = 0..15).
// ---------------------------------------------------------------------------
__global__ void packW_kernel(const float* __restrict__ W,
                             __bf16* __restrict__ Bp, int K, int Nout) {
    int idx = blockIdx.x * blockDim.x + threadIdx.x;
    int KB = K >> 5, NB = Nout >> 4;
    if (idx >= NB * KB * 32) return;
    int lane = idx & 31;
    int kb   = (idx >> 5) % KB;
    int nb   = (idx >> 5) / KB;
    int n     = nb * 16 + (lane & 15);
    int kbase = kb * 32 + (lane >> 4) * 16;
    __bf16* o = Bp + (size_t)idx * 16;
#pragma unroll
    for (int j = 0; j < 16; ++j)
        o[j] = to_bf16(W[(size_t)(kbase + j) * Nout + n]);
}

// ---------------------------------------------------------------------------
// WMMA GEMM: C(MxNout) = A(MxK, bf16 row-major) * B(packed fragments)
// One wave computes a 16x64 strip (4 WMMA accumulators, A fragment reused 4x).
// EPILOGUE 0: plain f32 store.  EPILOGUE 1: +bias, ReLU, bf16 store.
// ---------------------------------------------------------------------------
template <int EPILOGUE>
__global__ void gemm_wmma_kernel(const __bf16* __restrict__ A,
                                 const __bf16* __restrict__ Bp,
                                 const float* __restrict__ bias,
                                 void* __restrict__ Cout,
                                 int M, int K, int Nout) {
    const int waveId = threadIdx.x >> 5;
    const int lane   = threadIdx.x & 31;
    const int nGroups = Nout >> 6;              // 64-wide column groups
    const long mTiles = M >> 4;
    long tile = (long)blockIdx.x * (blockDim.x >> 5) + waveId;
    if (tile >= mTiles * nGroups) return;
    const int mt = (int)(tile / nGroups);
    const int ng = (int)(tile % nGroups);

    const int m  = lane & 15;
    const int kh = lane >> 4;
    const int KB = K >> 5;

    v8f acc[4] = {};
    const __bf16* arow = A + (size_t)(mt * 16 + m) * K + kh * 16;

    for (int kb = 0; kb < KB; ++kb) {
        ABFrag a;
        a.h[0] = *(const v8bf*)(arow + kb * 32);       // K = kb*32 + kh*16 + 0..7
        a.h[1] = *(const v8bf*)(arow + kb * 32 + 8);   // K = kb*32 + kh*16 + 8..15
#pragma unroll
        for (int t = 0; t < 4; ++t) {
            const int nb = ng * 4 + t;
            const v16bf bfrag =
                *(const v16bf*)(Bp + (((size_t)nb * KB + kb) * 32 + lane) * 16);
            acc[t] = __builtin_amdgcn_wmma_f32_16x16x32_bf16(
                false, a.v, false, bfrag, (short)0, acc[t], false, false);
        }
    }

    // C/D layout: n = lane&15 within 16-col tile, row = mt*16 + (lane>>4)*8 + r
    const int n0 = lane & 15;
    const int mh = lane >> 4;
#pragma unroll
    for (int t = 0; t < 4; ++t) {
        const int ncol = (ng * 4 + t) * 16 + n0;
#pragma unroll
        for (int r = 0; r < 8; ++r) {
            size_t idx = (size_t)(mt * 16 + mh * 8 + r) * Nout + ncol;
            float val = acc[t][r];
            if (EPILOGUE == 0) {
                ((float*)Cout)[idx] = val;
            } else {
                val += bias[ncol];
                val = val > 0.0f ? val : 0.0f;
                ((__bf16*)Cout)[idx] = to_bf16(val);
            }
        }
    }
}

// ---------------------------------------------------------------------------
// Buffer init kernels
// ---------------------------------------------------------------------------
__global__ void fill_zero_kernel(float* __restrict__ p, size_t n4) {
    size_t i = (size_t)blockIdx.x * blockDim.x + threadIdx.x;
    if (i < n4) *(float4*)(p + i * 4) = make_float4(0.f, 0.f, 0.f, 0.f);
}

__global__ void init_bias_rows_kernel(float* __restrict__ p,
                                      const float* __restrict__ bias,
                                      int F, size_t total) {
    size_t i = (size_t)blockIdx.x * blockDim.x + threadIdx.x;
    if (i < total) p[i] = bias[i & (size_t)(F - 1)];
}

__global__ void bias_relu_kernel(float* __restrict__ h,
                                 const float* __restrict__ bias,
                                 int F, size_t total) {
    size_t i = (size_t)blockIdx.x * blockDim.x + threadIdx.x;
    if (i >= total) return;
    float v = h[i] + bias[i & (size_t)(F - 1)];
    h[i] = v > 0.0f ? v : 0.0f;
}

// ---------------------------------------------------------------------------
// Atomic SpMM: one wave32 per (edge or self-loop).
// dst[row] += wn[e] * src[col]; features fit L2 (102 MB < 192 MB).
// ---------------------------------------------------------------------------
template <int F>
__global__ void spmm_atomic_kernel(const long long* __restrict__ rows,
                                   const long long* __restrict__ cols,
                                   const float* __restrict__ wn,
                                   const float* __restrict__ src,
                                   float* __restrict__ dst,
                                   int E, int N) {
    const long gwave = ((long)blockIdx.x * blockDim.x + threadIdx.x) >> 5;
    const int lane = threadIdx.x & 31;
    if (gwave >= (long)E + N) return;
    int r, c;
    if (gwave < E) { r = (int)rows[gwave]; c = (int)cols[gwave]; }
    else           { r = c = (int)(gwave - E); }
    const float w = wn[gwave];
    const float4* s = (const float4*)(src + (size_t)c * F);
    float* d = dst + (size_t)r * F;
#pragma unroll
    for (int j = 0; j < F / 128; ++j) {
        float4 v = s[j * 32 + lane];
        int base = (j * 32 + lane) * 4;
        atomicAdd(d + base + 0, w * v.x);
        atomicAdd(d + base + 1, w * v.y);
        atomicAdd(d + base + 2, w * v.z);
        atomicAdd(d + base + 3, w * v.w);
    }
}

// ---------------------------------------------------------------------------
// Host-side orchestration
// ---------------------------------------------------------------------------
static inline size_t alignup(size_t x) { return (x + 255) & ~(size_t)255; }

extern "C" void kernel_launch(void* const* d_in, const int* in_sizes, int n_in,
                              void* d_out, int out_size, void* d_ws, size_t ws_size,
                              hipStream_t stream) {
    const float*     x  = (const float*)d_in[0];
    const long long* ei = (const long long*)d_in[1];
    const float*     W1 = (const float*)d_in[2];
    const float*     b1 = (const float*)d_in[3];
    const float*     W2 = (const float*)d_in[4];
    const float*     b2 = (const float*)d_in[5];
    const float*     W3 = (const float*)d_in[6];
    const float*     b3 = (const float*)d_in[7];
    float* out = (float*)d_out;

    const int N = in_sizes[0] / 512;   // 100000 (divisible by 16)
    const int E = in_sizes[1] / 2;     // 3200000
    const long long* rows = ei;
    const long long* cols = ei + E;

    // workspace layout
    char* ws = (char*)d_ws;
    size_t off = 0;
    auto take = [&](size_t bytes) { char* p = ws + off; off += alignup(bytes); return p; };
    float*  deg  = (float*) take((size_t)N * 4);
    float*  wn   = (float*) take(((size_t)E + N) * 4);
    __bf16* W1p  = (__bf16*)take((size_t)512 * 256 * 2);
    __bf16* W2p  = (__bf16*)take((size_t)256 * 256 * 2);
    __bf16* W3p  = (__bf16*)take((size_t)256 * 128 * 2);
    __bf16* xbf  = (__bf16*)take((size_t)N * 512 * 2);
    float*  bufA = (float*) take((size_t)N * 256 * 4);   // t1 / a2 / t3 (aliased)
    float*  h1   = (float*) take((size_t)N * 256 * 4);
    __bf16* a2bf = (__bf16*)take((size_t)N * 256 * 2);
    __bf16* h2bf = (__bf16*)take((size_t)N * 256 * 2);
    (void)ws_size; (void)n_in; (void)out_size;

    const int T = 256;
    auto blks = [](long work, int t) { return (int)((work + t - 1) / t); };

    // 1) normalized adjacency weights
    init_deg_kernel<<<blks(N, T), T, 0, stream>>>(deg, N);
    count_deg_kernel<<<blks(E, T), T, 0, stream>>>(rows, deg, E);
    edge_weight_kernel<<<blks((long)E + N, T), T, 0, stream>>>(rows, cols, deg, wn, E, N);

    // 2) convert x to bf16; pack weights into WMMA B-fragment layout
    cvt_bf16_kernel<<<blks((size_t)N * 512 / 4, T), T, 0, stream>>>(x, xbf, (size_t)N * 512 / 4);
    packW_kernel<<<blks((256 / 16) * (512 / 32) * 32, T), T, 0, stream>>>(W1, W1p, 512, 256);
    packW_kernel<<<blks((256 / 16) * (256 / 32) * 32, T), T, 0, stream>>>(W2, W2p, 256, 256);
    packW_kernel<<<blks((128 / 16) * (256 / 32) * 32, T), T, 0, stream>>>(W3, W3p, 256, 128);

    // 3) layer 1: t1 = x @ W1  (WMMA)
    {
        long waves = (long)(N / 16) * (256 / 64);
        gemm_wmma_kernel<0><<<blks(waves, 8), T, 0, stream>>>(xbf, W1p, nullptr, bufA, N, 512, 256);
    }
    //    h1 = relu(spmm(t1) + b1)
    fill_zero_kernel<<<blks((size_t)N * 256 / 4, T), T, 0, stream>>>(h1, (size_t)N * 256 / 4);
    spmm_atomic_kernel<256><<<blks(((long)E + N) * 32, T), T, 0, stream>>>(rows, cols, wn, bufA, h1, E, N);
    bias_relu_kernel<<<blks((size_t)N * 256, T), T, 0, stream>>>(h1, b1, 256, (size_t)N * 256);

    // 4) layer 2: a2 = spmm(h1); h2 = relu(a2 @ W2 + b2) -> bf16
    fill_zero_kernel<<<blks((size_t)N * 256 / 4, T), T, 0, stream>>>(bufA, (size_t)N * 256 / 4);
    spmm_atomic_kernel<256><<<blks(((long)E + N) * 32, T), T, 0, stream>>>(rows, cols, wn, h1, bufA, E, N);
    cvt_bf16_kernel<<<blks((size_t)N * 256 / 4, T), T, 0, stream>>>(bufA, a2bf, (size_t)N * 256 / 4);
    {
        long waves = (long)(N / 16) * (256 / 64);
        gemm_wmma_kernel<1><<<blks(waves, 8), T, 0, stream>>>(a2bf, W2p, b2, h2bf, N, 256, 256);
    }

    // 5) layer 3: t3 = h2 @ W3; out = spmm(t3) + b3
    {
        long waves = (long)(N / 16) * (128 / 64);
        gemm_wmma_kernel<0><<<blks(waves, 8), T, 0, stream>>>(h2bf, W3p, nullptr, bufA, N, 256, 128);
    }
    init_bias_rows_kernel<<<blks((size_t)N * 128, T), T, 0, stream>>>(out, b3, 128, (size_t)N * 128);
    spmm_atomic_kernel<128><<<blks(((long)E + N) * 32, T), T, 0, stream>>>(rows, cols, wn, bufA, out, E, N);
}
